// HGNNModel_51908974739852
// MI455X (gfx1250) — compile-verified, matrix operand order checked by reading, not run on hardware
//
#include <hip/hip_runtime.h>
#include <hip/hip_bf16.h>

// ---------------------------------------------------------------------------
// HGNN forward for MI455X (gfx1250, wave32).
// Dense GEMMs use v_wmma_f32_16x16x32_f16 with operands pre-swizzled into the
// CDNA5 per-lane WMMA register layout; sparse incidence ops use f32 global
// atomics; pooling uses block reductions.
// ---------------------------------------------------------------------------

typedef _Float16 half_t;
typedef _Float16 v16h __attribute__((ext_vector_type(16)));
typedef float    v8f  __attribute__((ext_vector_type(8)));

namespace cfg {
constexpr int B    = 64;
constexpr int N    = 512;
constexpr int NV   = 32768;     // B*N
constexpr int NE   = 8192;
constexpr int NNZ  = 262144;
constexpr int SNNZ = 24576;
constexpr int D    = 300;
constexpr int HID  = 256;
constexpr int NC   = 10;
}

// ---------------- embedding gather: h0[v,d] = emb[x[v],d] -------------------
__global__ void k_embed(const int* __restrict__ x, const float* __restrict__ emb,
                        float* __restrict__ h0) {
  int i = blockIdx.x * blockDim.x + threadIdx.x;
  if (i >= cfg::NV * cfg::D) return;
  int v = i / cfg::D, d = i - v * cfg::D;
  h0[i] = emb[(size_t)x[v] * cfg::D + d];
}

// ---------------- sparse gather + atomic scatter ----------------------------
// dst[sidx[i], :] += vals[i] * src[gidx[i], :]
__global__ void k_spmm(const float* __restrict__ src, float* __restrict__ dst,
                       const int* __restrict__ gidx, const int* __restrict__ sidx,
                       const float* __restrict__ vals, int dim) {
  int i = blockIdx.x;
  int g = gidx[i], s = sidx[i];
  float v = vals[i];
  const float* srow = src + (size_t)g * dim;
  float*       drow = dst + (size_t)s * dim;
  for (int d = threadIdx.x; d < dim; d += blockDim.x)
    atomicAdd(&drow[d], v * srow[d]);
}

// ---------------- row scaling: h[r,:] *= deg[r] -----------------------------
__global__ void k_scale(float* __restrict__ h, const float* __restrict__ deg,
                        int rows, int dim) {
  int i = blockIdx.x * blockDim.x + threadIdx.x;
  if (i >= rows * dim) return;
  h[i] *= deg[i / dim];
}

// ---------------- WMMA operand packing --------------------------------------
// CDNA5 16-bit A (16x32) per-lane layout:
//   lanes 0-15 : halves 0..7 -> K 0..7,  halves 8..15 -> K 16..23   (row = lane)
//   lanes 16-31: halves 0..7 -> K 8..15, halves 8..15 -> K 24..31   (row = lane-16)
// k_local = (h&7) + ((h>>3)<<4) + ((lane>>4)<<3)
// Packed layout: [tile][kstep][lane(32)][half(16)]  (contiguous b128 loads)
__global__ void k_pack_A(const float* __restrict__ src, half_t* __restrict__ Ap,
                         int K, int ksteps) {
  int i = blockIdx.x * blockDim.x + threadIdx.x;
  int total = (cfg::NV / 16) * ksteps * 512;
  if (i >= total) return;
  int h    = i & 15;
  int lane = (i >> 4) & 31;
  int rest = i >> 9;                 // tm*ksteps + ks
  int ks   = rest % ksteps;
  int tm   = rest / ksteps;
  int row  = tm * 16 + (lane & 15);
  int k    = ks * 32 + (h & 7) + ((h >> 3) << 4) + ((lane >> 4) << 3);
  Ap[i] = (k < K) ? (half_t)src[(size_t)row * K + k] : (half_t)0.f;
}

// B (32x16) per-lane layout: lane%16 -> column N, same K mapping as A.
__global__ void k_pack_B(const float* __restrict__ W, half_t* __restrict__ Bp,
                         int K, int ksteps) {
  int i = blockIdx.x * blockDim.x + threadIdx.x;
  int total = (cfg::HID / 16) * ksteps * 512;
  if (i >= total) return;
  int h    = i & 15;
  int lane = (i >> 4) & 31;
  int rest = i >> 9;
  int ks   = rest % ksteps;
  int tn   = rest / ksteps;
  int col  = tn * 16 + (lane & 15);
  int k    = ks * 32 + (h & 7) + ((h >> 3) << 4) + ((lane >> 4) << 3);
  Bp[i] = (k < K) ? (half_t)W[(size_t)k * cfg::HID + col] : (half_t)0.f;
}

// ---------------- WMMA GEMM: out = relu(A @ B + bias) -----------------------
// grid = (NV/16, 2), block = 256 (8 wave32s). All 8 waves share one M-tile of
// A (staged in LDS); wave w computes N-tile blockIdx.y*8+w.
__global__ void __launch_bounds__(256) k_gemm(
    const half_t* __restrict__ Ap, const half_t* __restrict__ Bp,
    const float* __restrict__ bias, float* __restrict__ out, int ksteps) {
  __shared__ __align__(32) half_t As[10 * 512];   // up to 10 K-steps
  const int tid   = threadIdx.x;
  const int wave  = tid >> 5;
  const int lane  = tid & 31;
  const int tileM = blockIdx.x;
  const int tileN = blockIdx.y * 8 + wave;

  // cooperative, fully-coalesced stage of the A tile into LDS
  const uint4* Ag = (const uint4*)(Ap + (size_t)tileM * ksteps * 512);
  uint4* Al = (uint4*)As;
  const int n128 = ksteps * 64;                   // ksteps*512 halves / 8
  for (int i = tid; i < n128; i += 256) Al[i] = Ag[i];

  const half_t* Bbase = Bp + (size_t)tileN * ksteps * 512;
  __builtin_prefetch(Bbase, 0, 1);                // global_prefetch_b8
  __syncthreads();

  v8f acc = {0.f, 0.f, 0.f, 0.f, 0.f, 0.f, 0.f, 0.f};
  for (int ks = 0; ks < ksteps; ++ks) {
    v16h a = *(const v16h*)(As + ks * 512 + lane * 16);
    v16h b = *(const v16h*)(Bbase + ks * 512 + lane * 16);
    acc = __builtin_amdgcn_wmma_f32_16x16x32_f16(
        /*neg_a=*/false, a, /*neg_b=*/false, b,
        /*c_mod=*/(short)0, acc, /*reuse_a=*/false, /*reuse_b=*/false);
  }

  // C/D layout: VGPR v -> row v + 8*(lane>=16); lane%16 -> col
  const int col   = tileN * 16 + (lane & 15);
  const int rbase = tileM * 16 + ((lane >> 4) << 3);
  const float bc  = bias[col];
#pragma unroll
  for (int v = 0; v < 8; ++v) {
    float r = acc[v] + bc;
    r = r > 0.f ? r : 0.f;
    out[(size_t)(rbase + v) * cfg::HID + col] = r;
  }
}

// ---------------- attention logits: e[v] = [h,tf] @ attW + attb -------------
__global__ void k_att(const float* __restrict__ h, int dim,
                      const float* __restrict__ tf, const float* __restrict__ aW,
                      const float* __restrict__ ab, float* __restrict__ e) {
  int v = blockIdx.x * blockDim.x + threadIdx.x;
  if (v >= cfg::NV) return;
  const float* row = h + (size_t)v * dim;
  float s = ab[0] + tf[2 * v] * aW[dim] + tf[2 * v + 1] * aW[dim + 1];
  for (int d = 0; d < dim; ++d) s += row[d] * aW[d];
  e[v] = s;
}

// ---------------- global max (2-stage) --------------------------------------
__global__ void k_max1(const float* __restrict__ e, float* __restrict__ part) {
  __shared__ float s[256];
  float m = -3.4e38f;
  for (int i = blockIdx.x * 256 + threadIdx.x; i < cfg::NV; i += gridDim.x * 256)
    m = fmaxf(m, e[i]);
  s[threadIdx.x] = m; __syncthreads();
  for (int o = 128; o > 0; o >>= 1) {
    if (threadIdx.x < o) s[threadIdx.x] = fmaxf(s[threadIdx.x], s[threadIdx.x + o]);
    __syncthreads();
  }
  if (threadIdx.x == 0) part[blockIdx.x] = s[0];
}
__global__ void k_max2(const float* __restrict__ part, int n, float* __restrict__ g) {
  __shared__ float s[64];
  s[threadIdx.x] = (threadIdx.x < n) ? part[threadIdx.x] : -3.4e38f;
  __syncthreads();
  for (int o = 32; o > 0; o >>= 1) {
    if (threadIdx.x < o) s[threadIdx.x] = fmaxf(s[threadIdx.x], s[threadIdx.x + o]);
    __syncthreads();
  }
  if (threadIdx.x == 0) g[0] = s[0];
}

__global__ void k_exp(const float* __restrict__ e, const float* __restrict__ gmax,
                      float* __restrict__ w) {
  int v = blockIdx.x * blockDim.x + threadIdx.x;
  if (v >= cfg::NV) return;
  w[v] = expf(e[v] - gmax[0]);
}

// per-graph sum of attention weights (graphs are contiguous 512-node blocks)
__global__ void k_rowsum(const float* __restrict__ w, float* __restrict__ rs) {
  __shared__ float s[256];
  int g = blockIdx.x;
  float a = 0.f;
  for (int i = threadIdx.x; i < cfg::N; i += 256) a += w[g * cfg::N + i];
  s[threadIdx.x] = a; __syncthreads();
  for (int o = 128; o > 0; o >>= 1) {
    if (threadIdx.x < o) s[threadIdx.x] += s[threadIdx.x + o];
    __syncthreads();
  }
  if (threadIdx.x == 0) rs[g] = s[0];
}

// weighted-avg + max pool -> p[g] = [pooled(dim), max(dim)]
__global__ void k_pool(const float* __restrict__ h, int dim,
                       const float* __restrict__ w, const float* __restrict__ rs,
                       float* __restrict__ p) {
  int g = blockIdx.x;
  int d = blockIdx.y * blockDim.x + threadIdx.x;
  if (d >= dim) return;
  const float* hb = h + (size_t)g * cfg::N * dim + d;
  const float* wb = w + g * cfg::N;
  float acc = 0.f, mx = -1e9f;
  for (int i = 0; i < cfg::N; ++i) {
    float x = hb[(size_t)i * dim];
    acc += wb[i] * x;
    mx = fmaxf(mx, x);
  }
  p[(size_t)g * 2 * dim + d]       = acc / (rs[g] + 1e-10f);
  p[(size_t)g * 2 * dim + dim + d] = mx;
}

// ---------------- final classifier ------------------------------------------
__global__ void k_final(const float* __restrict__ p0, const float* __restrict__ p1,
                        const float* __restrict__ p2,
                        const float* __restrict__ pW0, const float* __restrict__ pb0,
                        const float* __restrict__ pW1, const float* __restrict__ pb1,
                        const float* __restrict__ pW2, const float* __restrict__ pb2,
                        float* __restrict__ out) {
  int i = blockIdx.x * blockDim.x + threadIdx.x;
  if (i >= cfg::B * cfg::NC) return;
  int g = i / cfg::NC, c = i - g * cfg::NC;
  float s = pb0[c] + pb1[c] + pb2[c];
  const float* q0 = p0 + (size_t)g * (2 * cfg::D);
  for (int k = 0; k < 2 * cfg::D; ++k)   s += q0[k] * pW0[k * cfg::NC + c];
  const float* q1 = p1 + (size_t)g * (2 * cfg::HID);
  for (int k = 0; k < 2 * cfg::HID; ++k) s += q1[k] * pW1[k * cfg::NC + c];
  const float* q2 = p2 + (size_t)g * (2 * cfg::HID);
  for (int k = 0; k < 2 * cfg::HID; ++k) s += q2[k] * pW2[k * cfg::NC + c];
  out[i] = s;
}

// ---------------------------------------------------------------------------
extern "C" void kernel_launch(void* const* d_in, const int* in_sizes, int n_in,
                              void* d_out, int out_size, void* d_ws, size_t ws_size,
                              hipStream_t stream) {
  (void)in_sizes; (void)n_in; (void)out_size; (void)ws_size;
  using namespace cfg;

  const int*   x         = (const int*)  d_in[0];
  const int*   inc_rows  = (const int*)  d_in[1];
  const int*   inc_cols  = (const int*)  d_in[2];
  const float* inc_vals  = (const float*)d_in[3];
  const int*   sent_rows = (const int*)  d_in[4];
  const int*   sent_cols = (const int*)  d_in[5];
  const float* sent_vals = (const float*)d_in[6];
  const float* deg_v     = (const float*)d_in[7];
  const float* deg_e     = (const float*)d_in[8];
  // d_in[9] graph_ids, d_in[10] max_pool_idx: structure (v/512) used directly
  const float* tf_idf    = (const float*)d_in[11];
  const float* emb       = (const float*)d_in[12];
  const float* W1 = (const float*)d_in[13]; const float* b1 = (const float*)d_in[14];
  const float* W2 = (const float*)d_in[15]; const float* b2 = (const float*)d_in[16];
  const float* attW0 = (const float*)d_in[17]; const float* attb0 = (const float*)d_in[18];
  const float* attW1 = (const float*)d_in[19]; const float* attb1 = (const float*)d_in[20];
  const float* attW2 = (const float*)d_in[21]; const float* attb2 = (const float*)d_in[22];
  const float* pW0 = (const float*)d_in[23]; const float* pb0 = (const float*)d_in[24];
  const float* pW1 = (const float*)d_in[25]; const float* pb1 = (const float*)d_in[26];
  const float* pW2 = (const float*)d_in[27]; const float* pb2 = (const float*)d_in[28];
  float* out = (float*)d_out;

  // ---- workspace carve-out (~187 MB total) ----
  size_t off = 0;
  auto alloc = [&](size_t bytes) -> void* {
    void* p = (char*)d_ws + off;
    off = (off + bytes + 255) & ~(size_t)255;
    return p;
  };
  float*  h0   = (float*) alloc((size_t)NV * D   * 4);
  float*  h1   = (float*) alloc((size_t)NV * HID * 4);
  float*  h2   = (float*) alloc((size_t)NV * HID * 4);
  float*  m    = (float*) alloc((size_t)NE * D   * 4);
  float*  m2   = (float*) alloc((size_t)NE * D   * 4);
  float*  hv   = (float*) alloc((size_t)NV * D   * 4);
  half_t* Ap   = (half_t*)alloc((size_t)(NV / 16) * 10 * 512 * 2);
  half_t* Bp   = (half_t*)alloc((size_t)(HID / 16) * 10 * 512 * 2);
  float*  ebuf = (float*) alloc((size_t)NV * 4);
  float*  wbuf = (float*) alloc((size_t)NV * 4);
  float*  part = (float*) alloc(64 * 4);
  float*  gmax = (float*) alloc(4);
  float*  rs   = (float*) alloc(B * 4);
  float*  p0   = (float*) alloc((size_t)B * 2 * D   * 4);
  float*  p1   = (float*) alloc((size_t)B * 2 * HID * 4);
  float*  p2   = (float*) alloc((size_t)B * 2 * HID * 4);

  auto run_layer = [&](const float* hin, int K, int ksteps, const float* W,
                       const float* bias, float* hout) {
    hipMemsetAsync(m, 0, (size_t)NE * K * 4, stream);
    k_spmm<<<NNZ, 128, 0, stream>>>(hin, m, inc_rows, inc_cols, inc_vals, K);
    k_scale<<<(NE * K + 255) / 256, 256, 0, stream>>>(m, deg_e, NE, K);
    hipMemsetAsync(m2, 0, (size_t)NE * K * 4, stream);
    k_spmm<<<SNNZ, 128, 0, stream>>>(m, m2, sent_rows, sent_cols, sent_vals, K);
    hipMemsetAsync(hv, 0, (size_t)NV * K * 4, stream);
    k_spmm<<<NNZ, 128, 0, stream>>>(m2, hv, inc_cols, inc_rows, inc_vals, K);
    k_scale<<<(NV * K + 255) / 256, 256, 0, stream>>>(hv, deg_v, NV, K);
    int totA = (NV / 16) * ksteps * 512;
    k_pack_A<<<(totA + 255) / 256, 256, 0, stream>>>(hv, Ap, K, ksteps);
    int totB = (HID / 16) * ksteps * 512;
    k_pack_B<<<(totB + 255) / 256, 256, 0, stream>>>(W, Bp, K, ksteps);
    dim3 g(NV / 16, (HID / 16) / 8);
    k_gemm<<<g, 256, 0, stream>>>(Ap, Bp, bias, hout, ksteps);
  };

  auto run_pool = [&](const float* h, int dim, const float* aW, const float* ab,
                      float* p) {
    k_att<<<(NV + 255) / 256, 256, 0, stream>>>(h, dim, tf_idf, aW, ab, ebuf);
    k_max1<<<64, 256, 0, stream>>>(ebuf, part);
    k_max2<<<1, 64, 0, stream>>>(part, 64, gmax);
    k_exp<<<(NV + 255) / 256, 256, 0, stream>>>(ebuf, gmax, wbuf);
    k_rowsum<<<B, 256, 0, stream>>>(wbuf, rs);
    dim3 g(B, (dim + 255) / 256);
    k_pool<<<g, 256, 0, stream>>>(h, dim, wbuf, rs, p);
  };

  // h0 = emb[x]
  k_embed<<<(NV * D + 255) / 256, 256, 0, stream>>>(x, emb, h0);

  // layer 1: K=300 -> Kpad=320 (10 k-steps of 32); layer 2: K=256 (8 k-steps)
  run_layer(h0, D,   10, W1, b1, h1);
  run_layer(h1, HID,  8, W2, b2, h2);

  run_pool(h0, D,   attW0, attb0, p0);
  run_pool(h1, HID, attW1, attb1, p1);
  run_pool(h2, HID, attW2, attb2, p2);

  k_final<<<(B * NC + 255) / 256, 256, 0, stream>>>(p0, p1, p2, pW0, pb0,
                                                    pW1, pb1, pW2, pb2, out);
}
